// DecoderAttention_38697655337107
// MI455X (gfx1250) — compile-verified, matrix operand order checked by reading, not run on hardware
//
#include <hip/hip_runtime.h>
#include <hip/hip_bf16.h>

typedef __bf16 bf16_t;
typedef __attribute__((ext_vector_type(16))) __bf16 v16bf;
typedef __attribute__((ext_vector_type(8)))  float  v8f;

#define HID_ 2048
#define BB_  2
#define TT_  2048
#define HH_  16
#define KVH_ 4
#define DD_  128
// combined qkv output columns: 2048 (q) + 512 (k) + 512 (v)
#define NQKV_ 3072

// ---------------------------------------------------------------------------
// f32 -> bf16 conversion (grid-stride)
// ---------------------------------------------------------------------------
__global__ void cvt_f32_bf16(const float* __restrict__ src,
                             bf16_t* __restrict__ dst, int n) {
  int i = blockIdx.x * blockDim.x + threadIdx.x;
  int stride = gridDim.x * blockDim.x;
  for (; i < n; i += stride) dst[i] = (bf16_t)src[i];
}

// ---------------------------------------------------------------------------
// Fused QKV projection + bias + RoPE.
// Block = one 128-col weight stripe shared by 8 waves; each wave owns a
// 16-row m-tile. All waves issue identical weight-fragment addresses each
// K-step, so the WGP$ serves 7/8 of the dominant (weight) traffic.
// Stripes 0..15 = Q heads, 16..19 = K heads, 20..23 = V heads
// (V written transposed [b,kvh,d,t] so PV B-operands are contiguous later).
// ---------------------------------------------------------------------------
__global__ void qkv_rope_kernel(const bf16_t* __restrict__ hsb,
                                const bf16_t* __restrict__ wb,
                                const float* __restrict__ bq,
                                const float* __restrict__ bk,
                                const float* __restrict__ bv,
                                const float* __restrict__ cosp,
                                const float* __restrict__ sinp,
                                bf16_t* __restrict__ qws,
                                bf16_t* __restrict__ kws,
                                bf16_t* __restrict__ vtws) {
  const int lane  = threadIdx.x & 31;
  const int w     = threadIdx.x >> 5;
  const int sidx  = blockIdx.x % 24;            // weight stripe (shared by block)
  const int mtile = (blockIdx.x / 24) * 8 + w;  // per-wave m-tile
  const int m0    = mtile * 16;
  const int c     = lane & 15;
  const int khalf = (lane < 16) ? 0 : 16;
  const int rg    = (lane >> 4) << 3;   // row-group offset for C layout

  v8f acc[8];
  #pragma unroll
  for (int j = 0; j < 8; ++j)
    #pragma unroll
    for (int i = 0; i < 8; ++i) acc[j][i] = 0.0f;

  const bf16_t* arow = hsb + (size_t)(m0 + c) * HID_ + khalf;
  const bf16_t* brow = wb + (size_t)(sidx * 128 + c) * HID_ + khalf;

  for (int k0 = 0; k0 < HID_; k0 += 32) {
    v16bf a = *(const v16bf*)(arow + k0);
    __builtin_prefetch(arow + k0 + 64, 0, 1);
    #pragma unroll
    for (int j = 0; j < 8; ++j) {
      v16bf bfr = *(const v16bf*)(brow + (size_t)j * 16 * HID_ + k0);
      acc[j] = __builtin_amdgcn_wmma_f32_16x16x32_bf16(
          false, a, false, bfr, (short)0, acc[j], false, false);
    }
  }

  // bias (column-indexed over combined qkv space)
  #pragma unroll
  for (int j = 0; j < 8; ++j) {
    int o = sidx * 128 + j * 16 + c;
    float bias = (o < 2048) ? bq[o] : (o < 2560 ? bk[o - 2048] : bv[o - 2560]);
    #pragma unroll
    for (int i = 0; i < 8; ++i) acc[j][i] += bias;
  }

  // RoPE for q and k stripes: pairs (d, d+64) live in acc[j], acc[j+4]
  if (sidx < 20) {
    #pragma unroll
    for (int i = 0; i < 8; ++i) {
      int grow = m0 + i + rg;
      int bb = grow >> 11;          // / TT_
      int tt = grow & (TT_ - 1);
      const float* cb = cosp + ((size_t)bb * TT_ + tt) * DD_;
      const float* sb = sinp + ((size_t)bb * TT_ + tt) * DD_;
      #pragma unroll
      for (int j = 0; j < 4; ++j) {
        int dlo = j * 16 + c;
        float cl = cb[dlo],      sl = sb[dlo];
        float ch = cb[dlo + 64], sh = sb[dlo + 64];
        float xlo = acc[j][i], xhi = acc[j + 4][i];
        acc[j][i]     = xlo * cl - xhi * sl;
        acc[j + 4][i] = xhi * ch + xlo * sh;
      }
    }
  }

  // store
  #pragma unroll
  for (int i = 0; i < 8; ++i) {
    int grow = m0 + i + rg;
    int bb = grow >> 11;
    int tt = grow & (TT_ - 1);
    #pragma unroll
    for (int j = 0; j < 8; ++j) {
      int d = j * 16 + c;
      bf16_t val = (bf16_t)acc[j][i];
      if (sidx < 16) {
        qws[(((size_t)bb * HH_ + sidx) * TT_ + tt) * DD_ + d] = val;
      } else if (sidx < 20) {
        kws[(((size_t)bb * KVH_ + (sidx - 16)) * TT_ + tt) * DD_ + d] = val;
      } else {
        vtws[(((size_t)bb * KVH_ + (sidx - 20)) * DD_ + d) * TT_ + tt] = val;
      }
    }
  }
}

// ---------------------------------------------------------------------------
// Flash attention (causal, GQA). One wave per (b, h, 16-row q-tile).
// Waves in a block have consecutive q-tiles of the same head, so they share
// the head's K/V stream through the WGP$.
// ---------------------------------------------------------------------------
__global__ void attn_kernel(const bf16_t* __restrict__ qws,
                            const bf16_t* __restrict__ kws,
                            const bf16_t* __restrict__ vtws,
                            bf16_t* __restrict__ aows) {
  __shared__ alignas(32) bf16_t plds[8][16][32];

  const int lane = threadIdx.x & 31;
  const int w    = threadIdx.x >> 5;
  const int gid  = blockIdx.x * 8 + w;
  const int per_b = HH_ * (TT_ / 16);
  const int b  = gid / per_b;
  const int r1 = gid - b * per_b;
  const int h  = r1 / (TT_ / 16);
  const int qt = r1 - h * (TT_ / 16);
  const int t0 = qt * 16;
  const int kvh = h / (HH_ / KVH_);
  const int c = lane & 15;
  const int khalf = (lane < 16) ? 0 : 16;
  const int rg = (lane >> 4) << 3;
  const float scale = 0.08838834764831845f;  // 1/sqrt(128)

  // Q fragments: 16x128, 4 k-steps of 32
  v16bf qf[4];
  const bf16_t* qrow = qws + (((size_t)b * HH_ + h) * TT_ + t0 + c) * DD_ + khalf;
  #pragma unroll
  for (int s = 0; s < 4; ++s) qf[s] = *(const v16bf*)(qrow + s * 32);

  v8f acc[8];
  #pragma unroll
  for (int j = 0; j < 8; ++j)
    #pragma unroll
    for (int i = 0; i < 8; ++i) acc[j][i] = 0.0f;
  float mrow[8], lrow[8];
  #pragma unroll
  for (int i = 0; i < 8; ++i) { mrow[i] = -1e30f; lrow[i] = 0.0f; }

  const bf16_t* kbase = kws + ((size_t)b * KVH_ + kvh) * TT_ * DD_;
  const bf16_t* vbase = vtws + ((size_t)b * KVH_ + kvh) * DD_ * TT_;

  for (int c0 = 0; c0 < t0 + 16; c0 += 32) {
    // S = Q x K^T for 32 kv columns (two 16-col tiles)
    v8f st[2];
    #pragma unroll
    for (int n = 0; n < 2; ++n) {
      v8f s;
      #pragma unroll
      for (int i = 0; i < 8; ++i) s[i] = 0.0f;
      const bf16_t* kr = kbase + (size_t)(c0 + n * 16 + c) * DD_ + khalf;
      #pragma unroll
      for (int ss = 0; ss < 4; ++ss) {
        v16bf kf = *(const v16bf*)(kr + ss * 32);
        s = __builtin_amdgcn_wmma_f32_16x16x32_bf16(
            false, qf[ss], false, kf, (short)0, s, false, false);
      }
      st[n] = s;
    }

    // scale + causal mask + online softmax
    float alpha[8];
    #pragma unroll
    for (int i = 0; i < 8; ++i) {
      int qrowi = t0 + i + rg;
      float rmax = -1e30f;
      #pragma unroll
      for (int n = 0; n < 2; ++n) {
        int tok = c0 + n * 16 + c;
        float v = st[n][i] * scale;
        if (tok > qrowi) v = -1e30f;
        st[n][i] = v;
        rmax = fmaxf(rmax, v);
      }
      #pragma unroll
      for (int off = 1; off < 16; off <<= 1)
        rmax = fmaxf(rmax, __shfl_xor(rmax, off, 16));
      float mnew = fmaxf(mrow[i], rmax);
      alpha[i] = __expf(mrow[i] - mnew);
      mrow[i] = mnew;
      float psum = 0.0f;
      #pragma unroll
      for (int n = 0; n < 2; ++n) {
        float p = __expf(st[n][i] - mnew);
        st[n][i] = p;
        psum += p;
      }
      #pragma unroll
      for (int off = 1; off < 16; off <<= 1)
        psum += __shfl_xor(psum, off, 16);
      lrow[i] = lrow[i] * alpha[i] + psum;
    }

    // P: C-layout -> row-major LDS -> A-layout fragment (same-wave, in-order DS)
    #pragma unroll
    for (int n = 0; n < 2; ++n)
      #pragma unroll
      for (int i = 0; i < 8; ++i)
        plds[w][i + rg][n * 16 + c] = (bf16_t)st[n][i];
    v16bf pa = *(const v16bf*)(&plds[w][c][khalf]);

    // acc = acc*alpha + P x V  (V stored transposed: contiguous per-lane rows)
    #pragma unroll
    for (int j = 0; j < 8; ++j) {
      #pragma unroll
      for (int i = 0; i < 8; ++i) acc[j][i] *= alpha[i];
      const bf16_t* vr = vbase + (size_t)(j * 16 + c) * TT_ + c0 + khalf;
      v16bf vf = *(const v16bf*)vr;
      acc[j] = __builtin_amdgcn_wmma_f32_16x16x32_bf16(
          false, pa, false, vf, (short)0, acc[j], false, false);
    }
  }

  // normalize and store [b, t, h*D + d] bf16
  #pragma unroll
  for (int i = 0; i < 8; ++i) {
    int tt = t0 + i + rg;
    float inv = 1.0f / lrow[i];
    #pragma unroll
    for (int j = 0; j < 8; ++j) {
      int d = j * 16 + c;
      aows[((size_t)b * TT_ + tt) * (HH_ * DD_) + h * DD_ + d] =
          (bf16_t)(acc[j][i] * inv);
    }
  }
}

// ---------------------------------------------------------------------------
// Output projection: out[bt, :] = attn_out[bt, :] x Wo^T  (f32 output).
// Same block mapping as qkv: waves in a block share the weight stripe.
// ---------------------------------------------------------------------------
__global__ void oproj_kernel(const bf16_t* __restrict__ aob,
                             const bf16_t* __restrict__ wob,
                             float* __restrict__ out) {
  const int lane  = threadIdx.x & 31;
  const int w     = threadIdx.x >> 5;
  const int sidx  = blockIdx.x % 16;            // weight stripe (shared by block)
  const int mtile = (blockIdx.x / 16) * 8 + w;  // per-wave m-tile
  const int m0    = mtile * 16;
  const int c     = lane & 15;
  const int khalf = (lane < 16) ? 0 : 16;
  const int rg    = (lane >> 4) << 3;

  v8f acc[8];
  #pragma unroll
  for (int j = 0; j < 8; ++j)
    #pragma unroll
    for (int i = 0; i < 8; ++i) acc[j][i] = 0.0f;

  const bf16_t* arow = aob + (size_t)(m0 + c) * HID_ + khalf;
  const bf16_t* brow = wob + (size_t)(sidx * 128 + c) * HID_ + khalf;

  for (int k0 = 0; k0 < HID_; k0 += 32) {
    v16bf a = *(const v16bf*)(arow + k0);
    __builtin_prefetch(arow + k0 + 64, 0, 1);
    #pragma unroll
    for (int j = 0; j < 8; ++j) {
      v16bf bfr = *(const v16bf*)(brow + (size_t)j * 16 * HID_ + k0);
      acc[j] = __builtin_amdgcn_wmma_f32_16x16x32_bf16(
          false, a, false, bfr, (short)0, acc[j], false, false);
    }
  }

  #pragma unroll
  for (int i = 0; i < 8; ++i) {
    int grow = m0 + i + rg;
    #pragma unroll
    for (int j = 0; j < 8; ++j)
      out[(size_t)grow * HID_ + sidx * 128 + j * 16 + c] = acc[j][i];
  }
}

// ---------------------------------------------------------------------------
// launch
// ---------------------------------------------------------------------------
extern "C" void kernel_launch(void* const* d_in, const int* in_sizes, int n_in,
                              void* d_out, int out_size, void* d_ws, size_t ws_size,
                              hipStream_t stream) {
  const float* hs   = (const float*)d_in[0];
  const float* cosp = (const float*)d_in[1];
  const float* sinp = (const float*)d_in[2];
  // d_in[3] = attention_mask (causal) — computed analytically in-kernel
  const float* Wq = (const float*)d_in[4];
  const float* bq = (const float*)d_in[5];
  const float* Wk = (const float*)d_in[6];
  const float* bk = (const float*)d_in[7];
  const float* Wv = (const float*)d_in[8];
  const float* bv = (const float*)d_in[9];
  const float* Wo = (const float*)d_in[10];
  float* out = (float*)d_out;

  char* ws = (char*)d_ws;
  bf16_t* hsb  = (bf16_t*)(ws + 0);           // 16 MB  (B*T x HID)
  bf16_t* wb   = (bf16_t*)(ws + 16777216);    // 12 MB  (3072 x 2048 qkv weights)
  bf16_t* wob  = (bf16_t*)(ws + 29360128);    //  8 MB  (2048 x 2048)
  bf16_t* qws  = (bf16_t*)(ws + 37748736);    // 16 MB  (B,H,T,D)
  bf16_t* kws  = (bf16_t*)(ws + 54525952);    //  4 MB  (B,KVH,T,D)
  bf16_t* vtws = (bf16_t*)(ws + 58720256);    //  4 MB  (B,KVH,D,T)
  bf16_t* aows = (bf16_t*)(ws + 62914560);    // 16 MB  (B*T x H*D)

  cvt_f32_bf16<<<2048, 256, 0, stream>>>(hs, hsb, BB_ * TT_ * HID_);
  cvt_f32_bf16<<<2048, 256, 0, stream>>>(Wq, wb, 2048 * 2048);
  cvt_f32_bf16<<<512, 256, 0, stream>>>(Wk, wb + (size_t)2048 * 2048, 512 * 2048);
  cvt_f32_bf16<<<512, 256, 0, stream>>>(Wv, wb + (size_t)2560 * 2048, 512 * 2048);
  cvt_f32_bf16<<<2048, 256, 0, stream>>>(Wo, wob, 2048 * 2048);

  // 24 stripes x 32 m-groups = 768 blocks; 8 waves (m-tiles) per block
  qkv_rope_kernel<<<768, 256, 0, stream>>>(hsb, wb, bq, bk, bv, cosp, sinp,
                                           qws, kws, vtws);
  // B*H*(T/16) = 4096 waves = 512 blocks
  attn_kernel<<<512, 256, 0, stream>>>(qws, kws, vtws, aows);
  // 16 stripes x 32 m-groups = 512 blocks; 8 waves per block
  oproj_kernel<<<512, 256, 0, stream>>>(aows, wob, out);
}
